// LinearAttentionLayer_70068096467988
// MI455X (gfx1250) — matrix operand order repaired and checked by reading, not compiled, not run on hardware
//
#include <hip/hip_runtime.h>
#include <hip/hip_bf16.h>

typedef __attribute__((ext_vector_type(16))) __bf16 v16bf;
typedef __attribute__((ext_vector_type(8)))  float  v8f;
typedef int v4i_t __attribute__((vector_size(16)));   // matches async-LDS builtin param type

#define D_DIM 1024
#define N_SEQ 4096
#define B_BAT 8
#define BM 128
#define BN 128
#define BK 32
#define BKP 40   // +8 bf16 pad -> 80B row stride: 16B aligned rows (b128 async), bank stride 20

#if defined(__has_builtin)
#if __has_builtin(__builtin_amdgcn_global_load_async_to_lds_b128)
#define ASYNC_LDS 1
#endif
#endif
#ifndef ASYNC_LDS
#define ASYNC_LDS 0
#endif

__device__ __forceinline__ void wait_async_lds() {
#if ASYNC_LDS
#if __has_builtin(__builtin_amdgcn_s_wait_asynccnt)
  __builtin_amdgcn_s_wait_asynccnt(0);
#else
  asm volatile("s_wait_asynccnt 0x0" ::: "memory");
#endif
#endif
}

// ---------------- LDS tile fill helpers (256 threads, 128x32 bf16 tiles) ----------------

// Direct copy: S[m][k] = src[m*stride + k]. One 32B chunk per thread.
__device__ __forceinline__ void fillD_async(__bf16 (&S)[BM][BKP], const __bf16* __restrict__ src,
                                            size_t stride, int tid) {
  const int m = tid >> 1;
  const int h = (tid & 1) * 16;                       // 16 bf16 = 32 bytes
  const __bf16* g = src + (size_t)m * stride + h;
#if ASYNC_LDS
  __builtin_amdgcn_global_load_async_to_lds_b128(
      (v4i_t*)(void*)g,       (v4i_t*)(void*)&S[m][h + 0], 0, 0);
  __builtin_amdgcn_global_load_async_to_lds_b128(
      (v4i_t*)(void*)(g + 8), (v4i_t*)(void*)&S[m][h + 8], 0, 0);
#else
  const uint4* g4 = reinterpret_cast<const uint4*>(g);
  *reinterpret_cast<uint4*>(&S[m][h + 0]) = g4[0];
  *reinterpret_cast<uint4*>(&S[m][h + 8]) = g4[1];
#endif
}

// fp32 source with convert: S[m][k] = (bf16) src[m*stride + k]
__device__ __forceinline__ void fillD_f32(__bf16 (&S)[BM][BKP], const float* __restrict__ src,
                                          size_t stride, int tid) {
#pragma unroll
  for (int i = 0; i < 4; ++i) {
    int idx = tid + i * 256;
    int m  = idx >> 3;
    int kg = idx & 7;
    float4 f = *reinterpret_cast<const float4*>(src + (size_t)m * stride + kg * 4);
    S[m][kg * 4 + 0] = (__bf16)f.x;
    S[m][kg * 4 + 1] = (__bf16)f.y;
    S[m][kg * 4 + 2] = (__bf16)f.z;
    S[m][kg * 4 + 3] = (__bf16)f.w;
  }
}

// Transposing fill: S[d][k] = src[k*stride + d]  (bf16 source; 32 k-rows, 128 d)
__device__ __forceinline__ void fillT_bf16(__bf16 (&S)[BM][BKP], const __bf16* __restrict__ src,
                                           size_t stride, int tid) {
#pragma unroll
  for (int i = 0; i < 8; ++i) {
    int idx = tid + i * 256;          // 2048 uint loads (pair of bf16 along d)
    int kn = idx >> 6;                // 0..31
    int dp = idx & 63;                // pair index, d = 2*dp
    unsigned u = *reinterpret_cast<const unsigned*>(src + (size_t)kn * stride + dp * 2);
    union { unsigned u; __bf16 h[2]; } c; c.u = u;
    S[dp * 2 + 0][kn] = c.h[0];
    S[dp * 2 + 1][kn] = c.h[1];
  }
}

// Prefetch next 32x128 bf16 tile (32 rows x 256B) into cache: one chunk per thread
__device__ __forceinline__ void prefetchT(const __bf16* src, int tid) {
  __builtin_prefetch(src + (size_t)(tid >> 3) * D_DIM + (size_t)(tid & 7) * 16, 0, 0);
}

// ---------------- WMMA fragment loads (ISA 7.12.2 bf16 layouts) ----------------

// A (16x32, MxK): lane L<16 -> row M=L, K in {0..7,16..23}; lane L+16 -> K in {8..15,24..31}
__device__ __forceinline__ v16bf load_frag_a(const __bf16 (&S)[BM][BKP], int mrow, int lane) {
  union { v16bf v; unsigned u[8]; } f;
  const int b0 = (lane < 16) ? 0 : 8;
  const int b1 = b0 + 16;
  const __bf16* row = &S[mrow][0];
#pragma unroll
  for (int j = 0; j < 4; ++j)
    f.u[j] = *reinterpret_cast<const unsigned*>(row + b0 + 2 * j);
#pragma unroll
  for (int j = 0; j < 4; ++j)
    f.u[4 + j] = *reinterpret_cast<const unsigned*>(row + b1 + 2 * j);
  return f.v;
}

// B (32x16, KxN): lane holds column n = L%16; lanes 0-15 K in {0..7,16..23}, lanes 16-31 K in {8..15,24..31}
__device__ __forceinline__ v16bf load_frag_b(const __bf16 (&S)[BN][BKP], int ncol, int lane) {
  union { v16bf v; unsigned u[8]; } f;
  const int kb = (lane < 16) ? 0 : 8;
  const __bf16* row = &S[ncol][0];
#pragma unroll
  for (int j = 0; j < 8; ++j)
    f.u[j] = *reinterpret_cast<const unsigned*>(row + kb + (j >> 2) * 16 + 2 * (j & 3));
  return f.v;
}

// ---------------- Kernel 0: fp32 -> bf16 pack (x and the four weight matrices) -------

__global__ __launch_bounds__(256) void cvt_bf16_kernel(const float* __restrict__ src,
                                                       __bf16* __restrict__ dst, int n4) {
  int idx = blockIdx.x * 256 + threadIdx.x;
  if (idx < n4) {
    float4 f = reinterpret_cast<const float4*>(src)[idx];
    union { uint2 u; __bf16 h[4]; } c;
    c.h[0] = (__bf16)f.x; c.h[1] = (__bf16)f.y;
    c.h[2] = (__bf16)f.z; c.h[3] = (__bf16)f.w;
    reinterpret_cast<uint2*>(dst)[idx] = c.u;
  }
}

// ---------------- Kernel 1: fused Q/K/V projections (x @ W.T + b, elu+1 on Q,K) ------

__global__ __launch_bounds__(256) void qkv_kernel(const __bf16* __restrict__ xbf,
    const __bf16* __restrict__ Wqb, const float* __restrict__ bq,
    const __bf16* __restrict__ Wkb, const float* __restrict__ bk,
    const __bf16* __restrict__ Wvb, const float* __restrict__ bv,
    __bf16* __restrict__ Q, __bf16* __restrict__ K, __bf16* __restrict__ V) {
  __shared__ __bf16 As[BM][BKP];
  __shared__ __bf16 Bs[BN][BKP];
  const int tid = threadIdx.x, lane = tid & 31, wave = tid >> 5;
  const int wm = (wave & 1) * 64, wn = (wave >> 1) * 32;
  const int m0 = blockIdx.x * BM;
  const int n0 = blockIdx.y * BN;
  const __bf16* W; const float* bias; __bf16* out; bool act;
  if (blockIdx.z == 0)      { W = Wqb; bias = bq; out = Q; act = true;  }
  else if (blockIdx.z == 1) { W = Wkb; bias = bk; out = K; act = true;  }
  else                      { W = Wvb; bias = bv; out = V; act = false; }

  v8f acc[4][2] = {};
  for (int k0 = 0; k0 < D_DIM; k0 += BK) {
    fillD_async(As, xbf + (size_t)m0 * D_DIM + k0, D_DIM, tid);
    fillD_async(Bs, W   + (size_t)n0 * D_DIM + k0, D_DIM, tid);
    wait_async_lds();
    __syncthreads();
    v16bf bfr[2];
#pragma unroll
    for (int ni = 0; ni < 2; ++ni)
      bfr[ni] = load_frag_b(Bs, wn + ni * 16 + (lane & 15), lane);
#pragma unroll
    for (int mi = 0; mi < 4; ++mi) {
      v16bf a = load_frag_a(As, wm + mi * 16 + (lane & 15), lane);
#pragma unroll
      for (int ni = 0; ni < 2; ++ni)
        acc[mi][ni] = __builtin_amdgcn_wmma_f32_16x16x32_bf16(
            false, a, false, bfr[ni], (short)0, acc[mi][ni], false, false);
    }
    __syncthreads();
  }
#pragma unroll
  for (int mi = 0; mi < 4; ++mi)
#pragma unroll
    for (int ni = 0; ni < 2; ++ni) {
      int col = n0 + wn + ni * 16 + (lane & 15);
      float bb = bias[col];
#pragma unroll
      for (int r = 0; r < 8; ++r) {
        int row = m0 + wm + mi * 16 + r + ((lane < 16) ? 0 : 8);
        float v = acc[mi][ni][r] + bb;
        if (act) v = fmaxf(v, 0.f) + __expf(fminf(v, 0.f));   // elu(v)+1, branchless
        out[(size_t)row * D_DIM + col] = (__bf16)v;
      }
    }
}

// ---------------- Kernel 2: per-batch KV = K^T @ V (k over sequence) ----------------

__global__ __launch_bounds__(256) void kv_kernel(const __bf16* __restrict__ K,
                                                 const __bf16* __restrict__ V,
                                                 __bf16* __restrict__ KV) {
  __shared__ __bf16 As[BM][BKP];
  __shared__ __bf16 Bs[BN][BKP];
  const int tid = threadIdx.x, lane = tid & 31, wave = tid >> 5;
  const int wm = (wave & 1) * 64, wn = (wave >> 1) * 32;
  const int b  = blockIdx.z;
  const int d0 = blockIdx.x * BM;
  const int e0 = blockIdx.y * BN;
  const __bf16* Kb = K + (size_t)b * N_SEQ * D_DIM;
  const __bf16* Vb = V + (size_t)b * N_SEQ * D_DIM;

  v8f acc[4][2] = {};
  for (int k0 = 0; k0 < N_SEQ; k0 += BK) {
    fillT_bf16(As, Kb + (size_t)k0 * D_DIM + d0, D_DIM, tid);  // As[d][n] = K[n, d]
    fillT_bf16(Bs, Vb + (size_t)k0 * D_DIM + e0, D_DIM, tid);  // Bs[e][n] = V[n, e]
    if (k0 + BK < N_SEQ) {
      prefetchT(Kb + (size_t)(k0 + BK) * D_DIM + d0, tid);
      prefetchT(Vb + (size_t)(k0 + BK) * D_DIM + e0, tid);
    }
    __syncthreads();
    v16bf bfr[2];
#pragma unroll
    for (int ni = 0; ni < 2; ++ni)
      bfr[ni] = load_frag_b(Bs, wn + ni * 16 + (lane & 15), lane);
#pragma unroll
    for (int mi = 0; mi < 4; ++mi) {
      v16bf a = load_frag_a(As, wm + mi * 16 + (lane & 15), lane);
#pragma unroll
      for (int ni = 0; ni < 2; ++ni)
        acc[mi][ni] = __builtin_amdgcn_wmma_f32_16x16x32_bf16(
            false, a, false, bfr[ni], (short)0, acc[mi][ni], false, false);
    }
    __syncthreads();
  }
  __bf16* KVb = KV + (size_t)b * D_DIM * D_DIM;
#pragma unroll
  for (int mi = 0; mi < 4; ++mi)
#pragma unroll
    for (int ni = 0; ni < 2; ++ni) {
      int col = e0 + wn + ni * 16 + (lane & 15);
#pragma unroll
      for (int r = 0; r < 8; ++r) {
        int row = d0 + wm + mi * 16 + r + ((lane < 16) ? 0 : 8);
        KVb[(size_t)row * D_DIM + col] = (__bf16)acc[mi][ni][r];
      }
    }
}

// ---------------- Kernel 3: Ksum[b,d] = sum_n K[b,n,d] + 1e-6 ----------------

__global__ __launch_bounds__(256) void ksum_kernel(const __bf16* __restrict__ K,
                                                   float* __restrict__ Ksum) {
  const int b = blockIdx.y;
  const int d = blockIdx.x * 256 + threadIdx.x;
  const __bf16* Kb = K + (size_t)b * N_SEQ * D_DIM;
  float s = 0.f;
  for (int n = 0; n < N_SEQ; ++n) s += (float)Kb[(size_t)n * D_DIM + d];
  Ksum[b * D_DIM + d] = s + 1e-6f;
}

// ---------------- Kernel 4: Z[b,n] = dot(Q[b,n,:], Ksum[b,:]) ----------------

__global__ __launch_bounds__(256) void z_kernel(const __bf16* __restrict__ Q,
                                                const float* __restrict__ Ksum,
                                                float* __restrict__ Z) {
  const int lane = threadIdx.x & 31, wave = threadIdx.x >> 5;
  const int row = blockIdx.x * 8 + wave;       // global row in [0, B*N)
  const int b = row >> 12;                     // N_SEQ == 4096
  const __bf16* q = Q + (size_t)row * D_DIM;
  const float* ks = Ksum + b * D_DIM;
  float s = 0.f;
  for (int d = lane; d < D_DIM; d += 32) s += (float)q[d] * ks[d];
#pragma unroll
  for (int off = 16; off > 0; off >>= 1) s += __shfl_xor(s, off, 32);
  if (lane == 0) Z[row] = s;
}

// ---------------- Kernel 5: ctx = (Q @ KV) / Z  (per batch) ----------------

__global__ __launch_bounds__(256) void ctx_kernel(const __bf16* __restrict__ Q,
                                                  const __bf16* __restrict__ KV,
                                                  const float* __restrict__ Z,
                                                  __bf16* __restrict__ C) {
  __shared__ __bf16 As[BM][BKP];
  __shared__ __bf16 Bs[BN][BKP];
  const int tid = threadIdx.x, lane = tid & 31, wave = tid >> 5;
  const int wm = (wave & 1) * 64, wn = (wave >> 1) * 32;
  const int b  = blockIdx.z;
  const int m0 = blockIdx.x * BM;              // row within batch
  const int e0 = blockIdx.y * BN;
  const __bf16* Qb  = Q  + (size_t)b * N_SEQ * D_DIM;
  const __bf16* KVb = KV + (size_t)b * D_DIM * D_DIM;
  const float*  Zb  = Z  + (size_t)b * N_SEQ;
  __bf16* Cb = C + (size_t)b * N_SEQ * D_DIM;

  v8f acc[4][2] = {};
  for (int k0 = 0; k0 < D_DIM; k0 += BK) {
    fillD_async(As, Qb + (size_t)m0 * D_DIM + k0, D_DIM, tid);
    fillT_bf16(Bs, KVb + (size_t)k0 * D_DIM + e0, D_DIM, tid); // Bs[e][d] = KV[d, e]
    if (k0 + BK < D_DIM) prefetchT(KVb + (size_t)(k0 + BK) * D_DIM + e0, tid);
    wait_async_lds();
    __syncthreads();
    v16bf bfr[2];
#pragma unroll
    for (int ni = 0; ni < 2; ++ni)
      bfr[ni] = load_frag_b(Bs, wn + ni * 16 + (lane & 15), lane);
#pragma unroll
    for (int mi = 0; mi < 4; ++mi) {
      v16bf a = load_frag_a(As, wm + mi * 16 + (lane & 15), lane);
#pragma unroll
      for (int ni = 0; ni < 2; ++ni)
        acc[mi][ni] = __builtin_amdgcn_wmma_f32_16x16x32_bf16(
            false, a, false, bfr[ni], (short)0, acc[mi][ni], false, false);
    }
    __syncthreads();
  }
#pragma unroll
  for (int mi = 0; mi < 4; ++mi)
#pragma unroll
    for (int ni = 0; ni < 2; ++ni) {
      int col = e0 + wn + ni * 16 + (lane & 15);
#pragma unroll
      for (int r = 0; r < 8; ++r) {
        int row = m0 + wm + mi * 16 + r + ((lane < 16) ? 0 : 8);
        float v = acc[mi][ni][r] / Zb[row];
        Cb[(size_t)row * D_DIM + col] = (__bf16)v;
      }
    }
}

// ---------------- Kernel 6: out = ctx @ Wo.T + bo  (fp32 output) ----------------

__global__ __launch_bounds__(256) void out_kernel(const __bf16* __restrict__ C,
                                                  const __bf16* __restrict__ Wob,
                                                  const float* __restrict__ bo,
                                                  float* __restrict__ out) {
  __shared__ __bf16 As[BM][BKP];
  __shared__ __bf16 Bs[BN][BKP];
  const int tid = threadIdx.x, lane = tid & 31, wave = tid >> 5;
  const int wm = (wave & 1) * 64, wn = (wave >> 1) * 32;
  const int m0 = blockIdx.x * BM;              // global row in [0, B*N)
  const int n0 = blockIdx.y * BN;

  v8f acc[4][2] = {};
  for (int k0 = 0; k0 < D_DIM; k0 += BK) {
    fillD_async(As, C   + (size_t)m0 * D_DIM + k0, D_DIM, tid);
    fillD_async(Bs, Wob + (size_t)n0 * D_DIM + k0, D_DIM, tid);
    wait_async_lds();
    __syncthreads();
    v16bf bfr[2];
#pragma unroll
    for (int ni = 0; ni < 2; ++ni)
      bfr[ni] = load_frag_b(Bs, wn + ni * 16 + (lane & 15), lane);
#pragma unroll
    for (int mi = 0; mi < 4; ++mi) {
      v16bf a = load_frag_a(As, wm + mi * 16 + (lane & 15), lane);
#pragma unroll
      for (int ni = 0; ni < 2; ++ni)
        acc[mi][ni] = __builtin_amdgcn_wmma_f32_16x16x32_bf16(
            false, a, false, bfr[ni], (short)0, acc[mi][ni], false, false);
    }
    __syncthreads();
  }
#pragma unroll
  for (int mi = 0; mi < 4; ++mi)
#pragma unroll
    for (int ni = 0; ni < 2; ++ni) {
      int col = n0 + wn + ni * 16 + (lane & 15);
      float bb = bo[col];
#pragma unroll
      for (int r = 0; r < 8; ++r) {
        int row = m0 + wm + mi * 16 + r + ((lane < 16) ? 0 : 8);
        out[(size_t)row * D_DIM + col] = acc[mi][ni][r] + bb;
      }
    }
}

// ---------------- launch ----------------

extern "C" void kernel_launch(void* const* d_in, const int* in_sizes, int n_in,
                              void* d_out, int out_size, void* d_ws, size_t ws_size,
                              hipStream_t stream) {
  (void)in_sizes; (void)n_in; (void)out_size; (void)ws_size;
  const float* x  = (const float*)d_in[0];
  const float* Wq = (const float*)d_in[1];
  const float* bq = (const float*)d_in[2];
  const float* Wk = (const float*)d_in[3];
  const float* bk = (const float*)d_in[4];
  const float* Wv = (const float*)d_in[5];
  const float* bv = (const float*)d_in[6];
  const float* Wo = (const float*)d_in[7];
  const float* bo = (const float*)d_in[8];
  float* out = (float*)d_out;

  char* ws = (char*)d_ws;
  const size_t XBYTES = (size_t)B_BAT * N_SEQ * D_DIM * sizeof(__bf16);  // 64 MB
  const size_t WBYTES = (size_t)D_DIM * D_DIM * sizeof(__bf16);          // 2 MB
  size_t off = 0;
  __bf16* xbf = (__bf16*)(ws + off); off += XBYTES;
  __bf16* Wqb = (__bf16*)(ws + off); off += WBYTES;
  __bf16* Wkb = (__bf16*)(ws + off); off += WBYTES;
  __bf16* Wvb = (__bf16*)(ws + off); off += WBYTES;
  __bf16* Wob = (__bf16*)(ws + off); off += WBYTES;
  __bf16* Q   = (__bf16*)(ws + off); off += XBYTES;
  __bf16* K   = (__bf16*)(ws + off); off += XBYTES;
  __bf16* V   = (__bf16*)(ws + off); off += XBYTES;
  __bf16* KV  = (__bf16*)(ws + off); off += (size_t)B_BAT * D_DIM * D_DIM * sizeof(__bf16);
  float*  Ksum= (float*) (ws + off); off += (size_t)B_BAT * D_DIM * sizeof(float);
  float*  Z   = (float*) (ws + off); off += (size_t)B_BAT * N_SEQ * sizeof(float);
  __bf16* Ctx = xbf;  // x is dead after qkv_kernel; reuse its 64 MB

  dim3 blk(256);
  const int XN4 = (B_BAT * N_SEQ * D_DIM) / 4;   // 8388608
  const int WN4 = (D_DIM * D_DIM) / 4;           // 262144
  cvt_bf16_kernel<<<dim3(XN4 / 256), blk, 0, stream>>>(x,  xbf, XN4);
  cvt_bf16_kernel<<<dim3(WN4 / 256), blk, 0, stream>>>(Wq, Wqb, WN4);
  cvt_bf16_kernel<<<dim3(WN4 / 256), blk, 0, stream>>>(Wk, Wkb, WN4);
  cvt_bf16_kernel<<<dim3(WN4 / 256), blk, 0, stream>>>(Wv, Wvb, WN4);
  cvt_bf16_kernel<<<dim3(WN4 / 256), blk, 0, stream>>>(Wo, Wob, WN4);

  qkv_kernel <<<dim3((B_BAT * N_SEQ) / BM, D_DIM / BN, 3), blk, 0, stream>>>(
      xbf, Wqb, bq, Wkb, bk, Wvb, bv, Q, K, V);
  kv_kernel  <<<dim3(D_DIM / BM, D_DIM / BN, B_BAT),       blk, 0, stream>>>(K, V, KV);
  ksum_kernel<<<dim3(D_DIM / 256, B_BAT),                  blk, 0, stream>>>(K, Ksum);
  z_kernel   <<<dim3((B_BAT * N_SEQ) / 8),                 blk, 0, stream>>>(Q, Ksum, Z);
  ctx_kernel <<<dim3(N_SEQ / BM, D_DIM / BN, B_BAT),       blk, 0, stream>>>(Q, KV, Z, Ctx);
  out_kernel <<<dim3((B_BAT * N_SEQ) / BM, D_DIM / BN),    blk, 0, stream>>>(Ctx, Wob, bo, out);
}
